// WanSelfAttention_40965398069723
// MI455X (gfx1250) — matrix operand order repaired and checked by reading, not compiled, not run on hardware
//
#include <hip/hip_runtime.h>
#include <cstdint>
#include <cstddef>

// ---------- types ----------
typedef __attribute__((ext_vector_type(16))) __bf16       v16bf;
typedef __attribute__((ext_vector_type(8)))  float        v8f;
typedef __attribute__((ext_vector_type(4)))  unsigned int u32x4;
typedef __attribute__((ext_vector_type(2)))  unsigned int u32x2;
typedef __attribute__((ext_vector_type(4)))  float        f32x4;
typedef __attribute__((ext_vector_type(4)))  int          v4i;

union FragU {
    u32x4 q[2];
    v16bf v;
};

__device__ __forceinline__ unsigned short f2bf(float f) {
    unsigned int u = __float_as_uint(f);
    u += 0x7FFFu + ((u >> 16) & 1u);   // round-to-nearest-even
    return (unsigned short)(u >> 16);
}
__device__ __forceinline__ unsigned int pack2(float a, float b) {
    return (unsigned int)f2bf(a) | ((unsigned int)f2bf(b) << 16);
}

// ---------- CDNA5 async Global->LDS copy (ASYNCcnt path) ----------
__device__ __forceinline__ void async_copy16(const void* gsrc, void* lds_dst) {
#if __has_builtin(__builtin_amdgcn_global_load_async_to_lds_b128)
    __builtin_amdgcn_global_load_async_to_lds_b128(
        (__attribute__((address_space(1))) v4i*)gsrc,
        (__attribute__((address_space(3))) v4i*)lds_dst, 0, 0);
#else
    asm volatile("global_load_async_to_lds_b128 %0, %1, off"
                 :
                 : "v"((__attribute__((address_space(3))) void*)lds_dst),
                   "v"(gsrc)
                 : "memory");
#endif
}
__device__ __forceinline__ void wait_async0() {
#if __has_builtin(__builtin_amdgcn_s_wait_asynccnt)
    __builtin_amdgcn_s_wait_asynccnt(0);
#else
    asm volatile("s_wait_asynccnt 0x0" ::: "memory");
#endif
}

// ---------- problem constants (reference: B=2,S=1024,DIM=3072,N=24,D=128) ----------
static constexpr int BB  = 2;
static constexpr int SS  = 1024;
static constexpr int DIM = 3072;
static constexpr int NH  = 24;
static constexpr int HD  = 128;

// =====================================================================
// fp32 -> bf16 bulk convert (streaming, non-temporal out)
// =====================================================================
__global__ __launch_bounds__(256) void cvt_f32_bf16(const float* __restrict__ in,
                                                    unsigned short* __restrict__ out,
                                                    int n) {
    int i = (blockIdx.x * 256 + threadIdx.x) * 4;
    if (i + 3 < n) {
        f32x4 v = *reinterpret_cast<const f32x4*>(in + i);
        u32x2 p;
        p.x = pack2(v.x, v.y);
        p.y = pack2(v.z, v.w);
        __builtin_nontemporal_store(p, reinterpret_cast<u32x2*>(out + i));
    }
}

// =====================================================================
// GEMM variant 1 (preferred): C = A(bf16,MxK) * B(bf16,NxK)^T + bias
// both tiles staged with async Global->LDS; hot loop is async+WMMA only
// =====================================================================
__global__ __launch_bounds__(256) void gemm_bb(const unsigned short* __restrict__ A,
                                               const unsigned short* __restrict__ Bw,
                                               const float* __restrict__ bias,
                                               float* __restrict__ C,
                                               int M, int N, int K) {
    __shared__ unsigned short As[2][128][40];
    __shared__ unsigned short Bs[2][128][40];

    const int tid  = threadIdx.x;
    const int lane = tid & 31;
    const int wave = tid >> 5;
    const int m0   = blockIdx.y * 128;
    const int n0   = blockIdx.x * 128;
    const int wm   = (wave >> 2) * 64;
    const int wn   = (wave & 3) * 32;
    const int lrow = lane & 15;
    const int khi  = (lane >> 4) * 8;
    const int kb   = (lane >> 4) * 16;

    auto stage = [&](int buf, int k0) {
#pragma unroll
        for (int t = 0; t < 2; ++t) {
            int u4   = tid * 2 + t;
            int row  = u4 >> 2;              // 4 x 16B per row (32 halves)
            int colh = (u4 & 3) * 8;
            async_copy16(&A[(size_t)(m0 + row) * K + k0 + colh], &As[buf][row][colh]);
            async_copy16(&Bw[(size_t)(n0 + row) * K + k0 + colh], &Bs[buf][row][colh]);
        }
    };

    v8f acc[4][2] = {};

    stage(0, 0);
    wait_async0();
    __syncthreads();

    for (int k0 = 0; k0 < K; k0 += 32) {
        const int buf = (k0 >> 5) & 1;
        if (k0 + 32 < K) stage(buf ^ 1, k0 + 32);   // async, overlaps WMMA below

        FragU af[4], bf[2];
#pragma unroll
        for (int mi = 0; mi < 4; ++mi) {
            af[mi].q[0] = *reinterpret_cast<const u32x4*>(&As[buf][wm + mi * 16 + lrow][khi]);
            af[mi].q[1] = *reinterpret_cast<const u32x4*>(&As[buf][wm + mi * 16 + lrow][16 + khi]);
        }
#pragma unroll
        for (int ni = 0; ni < 2; ++ni) {
            bf[ni].q[0] = *reinterpret_cast<const u32x4*>(&Bs[buf][wn + ni * 16 + lrow][khi]);
            bf[ni].q[1] = *reinterpret_cast<const u32x4*>(&Bs[buf][wn + ni * 16 + lrow][16 + khi]);
        }
#pragma unroll
        for (int mi = 0; mi < 4; ++mi)
#pragma unroll
            for (int ni = 0; ni < 2; ++ni)
                acc[mi][ni] = __builtin_amdgcn_wmma_f32_16x16x32_bf16(
                    false, af[mi].v, false, bf[ni].v, (short)0, acc[mi][ni], false, false);

        wait_async0();
        __syncthreads();
    }

#pragma unroll
    for (int mi = 0; mi < 4; ++mi) {
#pragma unroll
        for (int ni = 0; ni < 2; ++ni) {
            int col   = n0 + wn + ni * 16 + lrow;
            float bb  = bias[col];
            int rbase = m0 + wm + mi * 16 + (lane >> 4) * 8;
#pragma unroll
            for (int g = 0; g < 8; ++g)
                __builtin_nontemporal_store(acc[mi][ni][g] + bb,
                                            &C[(size_t)(rbase + g) * N + col]);
        }
    }
}

// =====================================================================
// GEMM variant 2 (fallback, small workspace): W kept fp32, converted in staging
// =====================================================================
__global__ __launch_bounds__(256) void gemm_xwT(const unsigned short* __restrict__ A,
                                                const float* __restrict__ W,
                                                const float* __restrict__ bias,
                                                float* __restrict__ C,
                                                int M, int N, int K) {
    __shared__ unsigned short As[2][128][40];
    __shared__ unsigned short Bs[2][128][40];

    const int tid  = threadIdx.x;
    const int lane = tid & 31;
    const int wave = tid >> 5;
    const int m0   = blockIdx.y * 128;
    const int n0   = blockIdx.x * 128;
    const int wm   = (wave >> 2) * 64;
    const int wn   = (wave & 3) * 32;
    const int lrow = lane & 15;
    const int khi  = (lane >> 4) * 8;
    const int kb   = (lane >> 4) * 16;

    auto stageA = [&](int buf, int k0) {
#pragma unroll
        for (int t = 0; t < 2; ++t) {
            int u4   = tid * 2 + t;
            int row  = u4 >> 2;
            int colh = (u4 & 3) * 8;
            async_copy16(&A[(size_t)(m0 + row) * K + k0 + colh], &As[buf][row][colh]);
        }
    };
    auto stageB = [&](int buf, int k0) {
#pragma unroll
        for (int t = 0; t < 4; ++t) {
            int f4   = tid * 4 + t;
            int row  = f4 >> 3;
            int colf = (f4 & 7) * 4;
            f32x4 v = *reinterpret_cast<const f32x4*>(&W[(size_t)(n0 + row) * K + k0 + colf]);
            u32x2 p;
            p.x = pack2(v.x, v.y);
            p.y = pack2(v.z, v.w);
            *reinterpret_cast<u32x2*>(&Bs[buf][row][colf]) = p;
        }
    };

    v8f acc[4][2] = {};

    stageA(0, 0);
    stageB(0, 0);
    wait_async0();
    __syncthreads();

    for (int k0 = 0; k0 < K; k0 += 32) {
        const int buf = (k0 >> 5) & 1;
        if (k0 + 32 < K) {
            __builtin_prefetch(&W[(size_t)(n0 + (tid >> 1)) * K + k0 + 64], 0, 0);
            stageA(buf ^ 1, k0 + 32);
            stageB(buf ^ 1, k0 + 32);
        }

        FragU af[4], bf[2];
#pragma unroll
        for (int mi = 0; mi < 4; ++mi) {
            af[mi].q[0] = *reinterpret_cast<const u32x4*>(&As[buf][wm + mi * 16 + lrow][khi]);
            af[mi].q[1] = *reinterpret_cast<const u32x4*>(&As[buf][wm + mi * 16 + lrow][16 + khi]);
        }
#pragma unroll
        for (int ni = 0; ni < 2; ++ni) {
            bf[ni].q[0] = *reinterpret_cast<const u32x4*>(&Bs[buf][wn + ni * 16 + lrow][khi]);
            bf[ni].q[1] = *reinterpret_cast<const u32x4*>(&Bs[buf][wn + ni * 16 + lrow][16 + khi]);
        }
#pragma unroll
        for (int mi = 0; mi < 4; ++mi)
#pragma unroll
            for (int ni = 0; ni < 2; ++ni)
                acc[mi][ni] = __builtin_amdgcn_wmma_f32_16x16x32_bf16(
                    false, af[mi].v, false, bf[ni].v, (short)0, acc[mi][ni], false, false);

        wait_async0();
        __syncthreads();
    }

#pragma unroll
    for (int mi = 0; mi < 4; ++mi) {
#pragma unroll
        for (int ni = 0; ni < 2; ++ni) {
            int col   = n0 + wn + ni * 16 + lrow;
            float bb  = bias[col];
            int rbase = m0 + wm + mi * 16 + (lane >> 4) * 8;
#pragma unroll
            for (int g = 0; g < 8; ++g)
                __builtin_nontemporal_store(acc[mi][ni][g] + bb,
                                            &C[(size_t)(rbase + g) * N + col]);
        }
    }
}

// =====================================================================
// RMS-norm (over DIM) + RoPE; emit Q,K -> [b][h][s][d] bf16, V -> [b][h][d][s] bf16
// =====================================================================
__global__ __launch_bounds__(256) void norm_rope(const float* __restrict__ qf,
                                                 const float* __restrict__ kf,
                                                 const float* __restrict__ vf,
                                                 const float* __restrict__ gq,
                                                 const float* __restrict__ gk,
                                                 const float* __restrict__ freqs,
                                                 unsigned short* __restrict__ Qb,
                                                 unsigned short* __restrict__ Kb,
                                                 unsigned short* __restrict__ Vt) {
    const int row  = blockIdx.x;
    const int b    = row >> 10;
    const int s    = row & 1023;
    const int tid  = threadIdx.x;
    const int lane = tid & 31;
    const int wave = tid >> 5;
    const float* qr = qf + (size_t)row * DIM;
    const float* kr = kf + (size_t)row * DIM;
    const float* vr = vf + (size_t)row * DIM;

    float sq = 0.f, sk = 0.f;
    for (int i = tid; i < DIM; i += 256) {
        float a = qr[i]; sq += a * a;
        float c = kr[i]; sk += c * c;
    }
#pragma unroll
    for (int off = 1; off < 32; off <<= 1) {
        sq += __shfl_xor(sq, off, 32);
        sk += __shfl_xor(sk, off, 32);
    }
    __shared__ float red[2][8];
    if (lane == 0) { red[0][wave] = sq; red[1][wave] = sk; }
    __syncthreads();
    float tq = 0.f, tk = 0.f;
#pragma unroll
    for (int w = 0; w < 8; ++w) { tq += red[0][w]; tk += red[1][w]; }
    const float rq = rsqrtf(tq / (float)DIM + 1e-6f);
    const float rk = rsqrtf(tk / (float)DIM + 1e-6f);

    for (int p = tid; p < DIM / 2; p += 256) {
        const int hh = p >> 6;
        const int i  = p & 63;
        const float ang = freqs[(size_t)s * 64 + i];
        float sn, cs;
        __sincosf(ang, &sn, &cs);
        const int c0 = hh * HD + 2 * i;
        const float qe = qr[c0] * rq * gq[c0];
        const float qo = qr[c0 + 1] * rq * gq[c0 + 1];
        const float ke = kr[c0] * rk * gk[c0];
        const float ko = kr[c0 + 1] * rk * gk[c0 + 1];
        const size_t base = ((size_t)(b * NH + hh) * SS + s) * HD + 2 * i;
        Qb[base]     = f2bf(qe * cs - qo * sn);
        Qb[base + 1] = f2bf(qe * sn + qo * cs);
        Kb[base]     = f2bf(ke * cs - ko * sn);
        Kb[base + 1] = f2bf(ke * sn + ko * cs);
        const size_t vb = ((size_t)(b * NH + hh) * HD + 2 * i) * SS + s;
        Vt[vb]      = f2bf(vr[c0]);
        Vt[vb + SS] = f2bf(vr[c0 + 1]);
    }
}

// =====================================================================
// flash attention: grid (S/128, NH, B); 8 waves x 16 query rows.
// =====================================================================
__global__ __launch_bounds__(256) void flash_attn(const unsigned short* __restrict__ Qb,
                                                  const unsigned short* __restrict__ Kb,
                                                  const unsigned short* __restrict__ Vt,
                                                  const int* __restrict__ seq_lens,
                                                  unsigned short* __restrict__ Ob) {
    const int b    = blockIdx.z;
    const int h    = blockIdx.y;
    const int tid  = threadIdx.x;
    const int lane = tid & 31;
    const int wave = tid >> 5;
    const int q0   = blockIdx.x * 128 + wave * 16;
    const int lrow = lane & 15;
    const int khi  = (lane >> 4) * 8;
    const int kb   = (lane >> 4) * 16;

    const unsigned short* Qh = Qb + (size_t)(b * NH + h) * SS * HD;
    const unsigned short* Kh = Kb + (size_t)(b * NH + h) * SS * HD;
    const unsigned short* Vh = Vt + (size_t)(b * NH + h) * HD * SS;
    const int seqlen = seq_lens[b];

    __shared__ unsigned short Ks[32][136];
    __shared__ unsigned short Vs[128][40];
    __shared__ unsigned short Pl[8][16][40];

    FragU qfr[4];
#pragma unroll
    for (int dk = 0; dk < 4; ++dk) {
        qfr[dk].q[0] = *reinterpret_cast<const u32x4*>(&Qh[(size_t)(q0 + lrow) * HD + dk * 32 + khi]);
        qfr[dk].q[1] = *reinterpret_cast<const u32x4*>(&Qh[(size_t)(q0 + lrow) * HD + dk * 32 + 16 + khi]);
    }

    v8f o[8] = {};
    float m_i[8], l_i[8];
#pragma unroll
    for (int g = 0; g < 8; ++g) { m_i[g] = -1e30f; l_i[g] = 0.f; }
    const float scale = 0.08838834764831845f;

    for (int j0 = 0; j0 < SS; j0 += 32) {
#pragma unroll
        for (int t = 0; t < 2; ++t) {            // K tile: 512 x 16B chunks
            int c   = tid * 2 + t;
            int row = c >> 4;
            int col = (c & 15) * 8;
            async_copy16(&Kh[(size_t)(j0 + row) * HD + col], &Ks[row][col]);
        }
#pragma unroll
        for (int t = 0; t < 2; ++t) {            // V tile: 512 x 16B chunks
            int c   = tid * 2 + t;
            int d   = c >> 2;
            int col = (c & 3) * 8;
            async_copy16(&Vh[(size_t)d * SS + j0 + col], &Vs[d][col]);
        }
        wait_async0();
        __syncthreads();

        v8f st[2] = {};
#pragma unroll
        for (int t = 0; t < 2; ++t) {
#pragma unroll
            for (int dk = 0; dk < 4; ++dk) {
                FragU kfr;
                kfr.q[0] = *reinterpret_cast<const u32x4*>(&Ks[t * 16 + lrow][dk * 32 + khi]);
                kfr.q[1] = *reinterpret_cast<const u32x4*>(&Ks[t * 16 + lrow][dk * 32 + 16 + khi]);
                st[t] = __builtin_amdgcn_wmma_f32_16x16x32_bf16(
                    false, qfr[dk].v, false, kfr.v, (short)0, st[t], false, false);
            }
        }
#pragma unroll
        for (int g = 0; g < 8; ++g) {
            float s0 = st[0][g] * scale;
            float s1 = st[1][g] * scale;
            if (j0 + lrow      >= seqlen) s0 = -1e30f;
            if (j0 + 16 + lrow >= seqlen) s1 = -1e30f;
            float r = fmaxf(s0, s1);
#pragma unroll
            for (int off = 1; off < 16; off <<= 1) r = fmaxf(r, __shfl_xor(r, off, 32));
            const float mnew  = fmaxf(m_i[g], r);
            const float alpha = __expf(m_i[g] - mnew);
            const float p0 = __expf(s0 - mnew);
            const float p1 = __expf(s1 - mnew);
            st[0][g] = p0;
            st[1][g] = p1;
            float rs = p0 + p1;
#pragma unroll
            for (int off = 1; off < 16; off <<= 1) rs += __shfl_xor(rs, off, 32);
            l_i[g] = l_i[g] * alpha + rs;
            m_i[g] = mnew;
#pragma unroll
            for (int ni = 0; ni < 8; ++ni) o[ni][g] *= alpha;
        }
#pragma unroll
        for (int g = 0; g < 8; ++g) {
            const int r = (lane >> 4) * 8 + g;
            Pl[wave][r][lrow]      = f2bf(st[0][g]);
            Pl[wave][r][16 + lrow] = f2bf(st[1][g]);
        }
        FragU pf;
        pf.q[0] = *reinterpret_cast<const u32x4*>(&Pl[wave][lrow][khi]);
        pf.q[1] = *reinterpret_cast<const u32x4*>(&Pl[wave][lrow][16 + khi]);
#pragma unroll
        for (int ni = 0; ni < 8; ++ni) {
            FragU vfr;
            vfr.q[0] = *reinterpret_cast<const u32x4*>(&Vs[ni * 16 + lrow][khi]);
            vfr.q[1] = *reinterpret_cast<const u32x4*>(&Vs[ni * 16 + lrow][16 + khi]);
            o[ni] = __builtin_amdgcn_wmma_f32_16x16x32_bf16(
                false, pf.v, false, vfr.v, (short)0, o[ni], false, false);
        }
        __syncthreads();
    }

#pragma unroll
    for (int ni = 0; ni < 8; ++ni) {
#pragma unroll
        for (int g = 0; g < 8; ++g) {
            const int qrow = q0 + (lane >> 4) * 8 + g;
            const float val = o[ni][g] / l_i[g];
            Ob[(size_t)(b * SS + qrow) * DIM + h * HD + ni * 16 + lrow] = f2bf(val);
        }
    }
}

// =====================================================================
// launch
// =====================================================================
extern "C" void kernel_launch(void* const* d_in, const int* in_sizes, int n_in,
                              void* d_out, int out_size, void* d_ws, size_t ws_size,
                              hipStream_t stream) {
    (void)in_sizes; (void)n_in; (void)out_size;
    const float* x        = (const float*)d_in[0];
    const int*   seq_lens = (const int*)d_in[1];
    /* d_in[2] grid_sizes: unused by the reference math */
    const float* freqs = (const float*)d_in[3];
    const float* Wq = (const float*)d_in[4];
    const float* bq = (const float*)d_in[5];
    const float* Wk = (const float*)d_in[6];
    const float* bk = (const float*)d_in[7];
    const float* Wv = (const float*)d_in[8];
    const float* bv = (const float*)d_in[9];
    const float* Wo = (const float*)d_in[10];
    const float* bo = (const float*)d_in[11];
    const float* gq = (const float*)d_in[12];
    const float* gk = (const float*)d_in[13];
    float* out = (float*)d_out;

    char* ws = (char*)d_ws;
    size_t off = 0;
    auto alloc = [&](size_t bytes) -> void* {
        void* p = ws + off;
        off += (bytes + 255) & ~(size_t)255;
        return p;
    };
    const size_t rows = (size_t)BB * SS;                               // 2048
    const size_t wsz  = (size_t)DIM * DIM;                             // 9.44M elems
    unsigned short* xbf = (unsigned short*)alloc(rows * DIM * 2);
    float*          qf  = (float*)alloc(rows * DIM * 4);
    float*          kf  = (float*)alloc(rows * DIM * 4);
    float*          vf  = (float*)alloc(rows * DIM * 4);
    unsigned short* Qb  = (unsigned short*)alloc(rows * DIM * 2);
    unsigned short* Kb  = (unsigned short*)alloc(rows * DIM * 2);
    unsigned short* Vt  = (unsigned short*)alloc(rows * DIM * 2);
    unsigned short* Ob  = (unsigned short*)alloc(rows * DIM * 2);
    unsigned short* Wqb = (unsigned short*)alloc(wsz * 2);             // bf16 weights
    unsigned short* Wkb = (unsigned short*)alloc(wsz * 2);
    unsigned short* Wvb = (unsigned short*)alloc(wsz * 2);
    unsigned short* Wob = (unsigned short*)alloc(wsz * 2);
    const bool big = (off <= ws_size);   // ws_size fixed -> deterministic path choice

    {
        const int n = (int)(rows * DIM);
        cvt_f32_bf16<<<dim3(n / (256 * 4)), 256, 0, stream>>>(x, xbf, n);
    }
    const dim3 gg(DIM / 128, (int)rows / 128);   // (24, 16)

    if (big) {
        const int nw = (int)wsz;
        cvt_f32_bf16<<<dim3(nw / 1024), 256, 0, stream>>>(Wq, Wqb, nw);
        cvt_f32_bf16<<<dim3(nw / 1024), 256, 0, stream>>>(Wk, Wkb, nw);
        cvt_f32_bf16<<<dim3(nw / 1024), 256, 0, stream>>>(Wv, Wvb, nw);
        cvt_f32_bf16<<<dim3(nw / 1024), 256, 0, stream>>>(Wo, Wob, nw);
        gemm_bb<<<gg, 256, 0, stream>>>(xbf, Wqb, bq, qf, (int)rows, DIM, DIM);
        gemm_bb<<<gg, 256, 0, stream>>>(xbf, Wkb, bk, kf, (int)rows, DIM, DIM);
        gemm_bb<<<gg, 256, 0, stream>>>(xbf, Wvb, bv, vf, (int)rows, DIM, DIM);
    } else {
        gemm_xwT<<<gg, 256, 0, stream>>>(xbf, Wq, bq, qf, (int)rows, DIM, DIM);
        gemm_xwT<<<gg, 256, 0, stream>>>(xbf, Wk, bk, kf, (int)rows, DIM, DIM);
        gemm_xwT<<<gg, 256, 0, stream>>>(xbf, Wv, bv, vf, (int)rows, DIM, DIM);
    }
    norm_rope<<<dim3((int)rows), 256, 0, stream>>>(qf, kf, vf, gq, gk, freqs, Qb, Kb, Vt);
    flash_attn<<<dim3(SS / 128, NH, BB), 256, 0, stream>>>(Qb, Kb, Vt, seq_lens, Ob);
    if (big) {
        gemm_bb<<<gg, 256, 0, stream>>>(Ob, Wob, bo, out, (int)rows, DIM, DIM);
    } else {
        gemm_xwT<<<gg, 256, 0, stream>>>(Ob, Wo, bo, out, (int)rows, DIM, DIM);
    }
}
